// PaiNNMessage_12489764897067
// MI455X (gfx1250) — matrix operand order repaired and checked by reading, not compiled, not run on hardware
//
#include <hip/hip_runtime.h>
#include <hip/hip_bf16.h>

typedef __attribute__((ext_vector_type(16))) _Float16 v16h;
typedef __attribute__((ext_vector_type(8)))  float    v8f;

#define NN 10000
#define EE 160000
#define CC 128
#define SCALE_F 0.0625f
#define EPS_F 1.1920929e-07f

// workspace layout (bytes):
//   [0, 15360000)          phi  : float[N][384]
//   [15360000, +32768)     w1f  : f16 B-fragments, 8 ntiles x 4 ksteps x 32 lanes x 16
//   [15392768, +98304)     w2f  : f16 B-fragments, 24 ntiles x 4 ksteps x 32 lanes x 16
#define PHI_OFF  0
#define W1F_OFF  15360000
#define W2F_OFF  15392768

static __device__ __forceinline__ void atomAddF(float* p, float v) {
    __hip_atomic_fetch_add(p, v, __ATOMIC_RELAXED, __HIP_MEMORY_SCOPE_AGENT);
}

// ---------------- kernel 0: swizzle weights into WMMA B-fragment order (f16) ---------
__global__ __launch_bounds__(256) void painn_swizzle_w(const float* __restrict__ w1,
                                                       const float* __restrict__ w2,
                                                       _Float16* __restrict__ w1f,
                                                       _Float16* __restrict__ w2f) {
    int gid = blockIdx.x * 256 + threadIdx.x;       // 65536 total
    int u = gid;
    const float* w = w1;
    _Float16* dst = w1f;
    if (gid >= 8 * 4 * 32 * 16) { u = gid - 8 * 4 * 32 * 16; w = w2; dst = w2f; }
    int t    = u & 15;
    int lane = (u >> 4) & 31;
    int ks   = (u >> 9) & 3;
    int nt   = u >> 11;
    int n = nt * 16 + (lane & 15);
    int K = ks * 32 + (t >> 3) * 16 + (lane >> 4) * 8 + (t & 7);
    dst[u] = (_Float16)w[n * CC + K];               // B[K][n] = W[n][K]
}

// ---------------- kernel 1: out = concat(s, v) ---------------------------------------
__global__ __launch_bounds__(256) void painn_init_out(const float* __restrict__ s,
                                                      const float* __restrict__ v,
                                                      float* __restrict__ out) {
    int gid = blockIdx.x * 256 + threadIdx.x;       // N*512 total
    if (gid < NN * CC) out[gid] = s[gid];
    else               out[gid] = v[gid - NN * CC];
}

// ---------------- kernel 2: node MLP via WMMA ----------------------------------------
// block = 128 threads (4 waves), 16 nodes per block; grid = N/16 = 625
__global__ __launch_bounds__(128) void painn_node_mlp(const float* __restrict__ s,
                                                      const float* __restrict__ rms_w,
                                                      const float* __restrict__ b1,
                                                      const float* __restrict__ b2,
                                                      const _Float16* __restrict__ w1f,
                                                      const _Float16* __restrict__ w2f,
                                                      float* __restrict__ phi) {
    __shared__ _Float16 sA[16][CC];   // RMS-normed s, f16
    __shared__ _Float16 sH[16][CC];   // hidden h, f16
    const int t = threadIdx.x;
    const int node0 = blockIdx.x * 16;

    // ---- RMSNorm: 8 threads per node, 16 channels each ----
    {
        int m = t >> 3, sub = t & 7;
        const float* srow = s + (size_t)(node0 + m) * CC + sub * 16;
        float vals[16]; float ss = 0.f;
        #pragma unroll
        for (int u = 0; u < 16; ++u) { vals[u] = srow[u]; ss += vals[u] * vals[u]; }
        ss += __shfl_xor(ss, 1, 8);
        ss += __shfl_xor(ss, 2, 8);
        ss += __shfl_xor(ss, 4, 8);
        float rs = rsqrtf(ss * (1.0f / CC) + EPS_F);
        #pragma unroll
        for (int u = 0; u < 16; ++u)
            sA[m][sub * 16 + u] = (_Float16)(vals[u] * rs * rms_w[sub * 16 + u]);
    }
    __syncthreads();

    const int wave = t >> 5, lane = t & 31;
    const int mrow = lane & 15, g = lane >> 4;

    // ---- GEMM1: h = silu(s_norm @ W1^T + b1); 8 N-tiles, 2 per wave ----
    #pragma unroll
    for (int ti = 0; ti < 2; ++ti) {
        int nt = wave * 2 + ti;
        v8f acc = {};
        #pragma unroll
        for (int ks = 0; ks < 4; ++ks) {
            v16h a, b;
            const _Float16* ar = &sA[mrow][ks * 32 + g * 8];
            #pragma unroll
            for (int u = 0; u < 8; ++u) { a[u] = ar[u]; a[8 + u] = ar[16 + u]; }
            b = *(const v16h*)(w1f + (((nt * 4 + ks) * 32 + lane) << 4));
            acc = __builtin_amdgcn_wmma_f32_16x16x32_f16(false, a, false, b,
                                                         (short)0, acc, false, false);
        }
        int n = nt * 16 + mrow;
        float bb = b1[n];
        #pragma unroll
        for (int r = 0; r < 8; ++r) {
            float x = acc[r] + bb;
            sH[g * 8 + r][n] = (_Float16)(x / (1.f + __expf(-x)));   // SiLU
        }
    }
    __syncthreads();

    // ---- GEMM2: phi = h @ W2^T + b2; 24 N-tiles, 6 per wave ----
    #pragma unroll
    for (int ti = 0; ti < 6; ++ti) {
        int nt = wave * 6 + ti;
        v8f acc = {};
        #pragma unroll
        for (int ks = 0; ks < 4; ++ks) {
            v16h a, b;
            const _Float16* ar = &sH[mrow][ks * 32 + g * 8];
            #pragma unroll
            for (int u = 0; u < 8; ++u) { a[u] = ar[u]; a[8 + u] = ar[16 + u]; }
            b = *(const v16h*)(w2f + (((nt * 4 + ks) * 32 + lane) << 4));
            acc = __builtin_amdgcn_wmma_f32_16x16x32_f16(false, a, false, b,
                                                         (short)0, acc, false, false);
        }
        int n = nt * 16 + mrow;
        float bb = b2[n];
        #pragma unroll
        for (int r = 0; r < 8; ++r)
            phi[(size_t)(node0 + g * 8 + r) * 384 + n] = acc[r] + bb;
    }
}

// ---------------- kernel 3: edge filter + atomic scatter -----------------------------
// 32 lanes per edge, 4 channels (float4) per lane; grid = E*32/256 = 20000
__global__ __launch_bounds__(256) void painn_edge(const int* __restrict__ ei,
                                                  const float* __restrict__ rbf,
                                                  const float* __restrict__ ev,
                                                  const float* __restrict__ vin,
                                                  const float* __restrict__ phi,
                                                  float* __restrict__ out) {
    int gid = blockIdx.x * 256 + threadIdx.x;
    int e = gid >> 5;
    if (e >= EE) return;
    int lane = threadIdx.x & 31;
    int i = ei[e];
    int j = ei[EE + e];

    const float4* phij = (const float4*)(phi + (size_t)j * 384);
    const float4* rbfe = (const float4*)(rbf + (size_t)e * 384);
    float4 p0 = phij[lane],      r0 = rbfe[lane];        // m_s
    float4 p1 = phij[32 + lane], r1 = rbfe[32 + lane];   // m_vv
    float4 p2 = phij[64 + lane], r2 = rbfe[64 + lane];   // m_vs
    float ms[4]  = { p0.x * r0.x, p0.y * r0.y, p0.z * r0.z, p0.w * r0.w };
    float mvv[4] = { p1.x * r1.x, p1.y * r1.y, p1.z * r1.z, p1.w * r1.w };
    float mvs[4] = { p2.x * r2.x, p2.y * r2.y, p2.z * r2.z, p2.w * r2.w };

    int c = lane * 4;
    float* outs = out + (size_t)i * CC + c;
    #pragma unroll
    for (int u = 0; u < 4; ++u) atomAddF(outs + u, ms[u] * SCALE_F);

    float evk[3] = { ev[e * 3 + 0], ev[e * 3 + 1], ev[e * 3 + 2] };
    const float* vj = vin + (size_t)j * 384 + c;
    float* outv = out + (size_t)NN * CC + (size_t)i * 384 + c;
    #pragma unroll
    for (int k = 0; k < 3; ++k) {
        float4 vk = *(const float4*)(vj + k * CC);
        float gk[4] = { vk.x * mvv[0] + evk[k] * mvs[0],
                        vk.y * mvv[1] + evk[k] * mvs[1],
                        vk.z * mvv[2] + evk[k] * mvs[2],
                        vk.w * mvv[3] + evk[k] * mvs[3] };
        #pragma unroll
        for (int u = 0; u < 4; ++u) atomAddF(outv + k * CC + u, gk[u] * SCALE_F);
    }
}

extern "C" void kernel_launch(void* const* d_in, const int* in_sizes, int n_in,
                              void* d_out, int out_size, void* d_ws, size_t ws_size,
                              hipStream_t stream) {
    const float* s     = (const float*)d_in[0];
    const float* v     = (const float*)d_in[1];
    const int*   ei    = (const int*)d_in[2];
    const float* rbf   = (const float*)d_in[3];
    const float* ev    = (const float*)d_in[4];
    const float* w1    = (const float*)d_in[5];
    const float* b1    = (const float*)d_in[6];
    const float* w2    = (const float*)d_in[7];
    const float* b2    = (const float*)d_in[8];
    const float* rms_w = (const float*)d_in[9];
    float* out = (float*)d_out;
    char*  ws  = (char*)d_ws;

    float*     phi = (float*)(ws + PHI_OFF);
    _Float16*  w1f = (_Float16*)(ws + W1F_OFF);
    _Float16*  w2f = (_Float16*)(ws + W2F_OFF);

    painn_swizzle_w<<<(8 * 4 * 32 * 16 + 24 * 4 * 32 * 16) / 256, 256, 0, stream>>>(w1, w2, w1f, w2f);
    painn_init_out<<<(NN * 4 * CC) / 256, 256, 0, stream>>>(s, v, out);
    painn_node_mlp<<<NN / 16, 128, 0, stream>>>(s, rms_w, b1, b2, w1f, w2f, phi);
    painn_edge<<<(EE * 32) / 256, 256, 0, stream>>>(ei, rbf, ev, v, phi, out);
}
